// MessagePassingLayer_28192165331134
// MI455X (gfx1250) — compile-verified, matrix operand order checked by reading, not run on hardware
//
#include <hip/hip_runtime.h>

// ---------------------------------------------------------------------------
// Message-passing GNN layer for MI455X (gfx1250), bf16 WMMA path, round 2.
//   - 2 M-tiles (32 rows) per wave: each LDS B fragment feeds 2 WMMAs.
//   - Software double-buffered B fragments to hide ds_load latency.
//   - global_prefetch of next edge tile (edges = the only HBM stream).
// ---------------------------------------------------------------------------

typedef __attribute__((ext_vector_type(16))) __bf16 v16bf;
typedef __attribute__((ext_vector_type(8)))  float  v8f;

#define FDIM 128
#define HDIM 128
#define K1   384   // 3*F   (edge MLP layer-1 K)
#define K2   256   // F+H   (node MLP layer-1 K)

static __device__ __forceinline__ float swishf(float x) {
    return x * (1.0f / (1.0f + __expf(-x)));
}

// Load 16 consecutive fp32 at base+ko, convert to a bf16 A/B fragment.
static __device__ __forceinline__ v16bf load_frag_g(const float* __restrict__ base, int ko) {
    float a[16];
    const float4* src = (const float4*)(base + ko);
    #pragma unroll
    for (int v = 0; v < 4; v++) {
        float4 f = src[v];
        a[4*v+0] = f.x; a[4*v+1] = f.y; a[4*v+2] = f.z; a[4*v+3] = f.w;
    }
    v16bf r;
    #pragma unroll
    for (int i = 0; i < 16; i++) r[i] = (__bf16)a[i];
    return r;
}

static __device__ __forceinline__ v16bf load_frag_lds(const __bf16* p) {
    return *(const v16bf*)p;
}

// ---------------------------------------------------------------------------
// Edge MLP + scatter-add.  8 waves/WG, each wave owns 32 edge rows (2 M-tiles).
// LDS: Wm1^T 96KB + Wm2^T 32KB + act scratch 64KB = 192KB.
// ---------------------------------------------------------------------------
__launch_bounds__(256, 1)
__global__ void edge_mlp_kernel(const float* __restrict__ nodes,
                                const float* __restrict__ edges,
                                const int*   __restrict__ senders,
                                const int*   __restrict__ receivers,
                                const float* __restrict__ Wm1,
                                const float* __restrict__ bm1,
                                const float* __restrict__ Wm2,
                                const float* __restrict__ bm2,
                                float*       __restrict__ agg,
                                int E, int n_tiles)
{
    __shared__ __align__(32) __bf16 Wm1T[HDIM][K1];       // 98304 B
    __shared__ __align__(32) __bf16 Wm2T[HDIM][HDIM];     // 32768 B
    __shared__ __align__(32) __bf16 act[8][32][HDIM];     // 65536 B

    const int tid = threadIdx.x;

    // Stage transposed bf16 weights once (persistent workgroups).
    for (int i = tid; i < K1 * HDIM; i += 256) {
        int k = i >> 7;
        int n = i & 127;
        Wm1T[n][k] = (__bf16)Wm1[i];
    }
    for (int i = tid; i < HDIM * HDIM; i += 256) {
        int k = i >> 7;
        int n = i & 127;
        Wm2T[n][k] = (__bf16)Wm2[i];
    }
    __syncthreads();

    const int lane = tid & 31;
    const int wave = tid >> 5;
    const int hi   = lane >> 4;    // K-half select (A/B 16-bit layout)
    const int l16  = lane & 15;

    for (int tile = blockIdx.x; tile < n_tiles; tile += gridDim.x) {
        const int ebase = tile * 256 + wave * 32;

        // Prefetch next tile's edge rows (the only real HBM stream).
        {
            const int tn = tile + gridDim.x;
            if (tn < n_tiles) {
                int en0 = tn * 256 + wave * 32 + l16;
                int en1 = en0 + 16;
                en0 = (en0 < E) ? en0 : (E - 1);
                en1 = (en1 < E) ? en1 : (E - 1);
                __builtin_prefetch(edges + (size_t)en0 * FDIM, 0, 1);
                __builtin_prefetch(edges + (size_t)en1 * FDIM, 0, 1);
            }
        }

        // Gather row pointers for both M-tiles owned by this wave.
        const int e0  = ebase + l16;
        const int e1  = ebase + 16 + l16;
        const int ec0 = (e0 < E) ? e0 : (E - 1);
        const int ec1 = (e1 < E) ? e1 : (E - 1);
        const float* p0a = nodes + (size_t)senders[ec0]   * FDIM;
        const float* p1a = nodes + (size_t)receivers[ec0] * FDIM;
        const float* p2a = edges + (size_t)ec0            * FDIM;
        const float* p0b = nodes + (size_t)senders[ec1]   * FDIM;
        const float* p1b = nodes + (size_t)receivers[ec1] * FDIM;
        const float* p2b = edges + (size_t)ec1            * FDIM;

        v8f acc0[8] = {};
        v8f acc1[8] = {};

        // ---- layer 1: [32,384] x [384,128] ----
        #pragma unroll
        for (int ks = 0; ks < 12; ks++) {
            const int k0 = ks * 32;
            const int ko = (k0 & 127) + hi * 16;
            const float* ba = (k0 < 128) ? p0a : ((k0 < 256) ? p1a : p2a);
            const float* bb = (k0 < 128) ? p0b : ((k0 < 256) ? p1b : p2b);
            v16bf a0 = load_frag_g(ba, ko);
            v16bf a1 = load_frag_g(bb, ko);

            v16bf bcur = load_frag_lds(&Wm1T[l16][k0 + hi * 16]);
            #pragma unroll
            for (int t = 0; t < 8; t++) {
                v16bf bnext = bcur;
                if (t < 7) bnext = load_frag_lds(&Wm1T[(t + 1) * 16 + l16][k0 + hi * 16]);
                acc0[t] = __builtin_amdgcn_wmma_f32_16x16x32_bf16(
                    false, a0, false, bcur, (short)0, acc0[t], false, false);
                acc1[t] = __builtin_amdgcn_wmma_f32_16x16x32_bf16(
                    false, a1, false, bcur, (short)0, acc1[t], false, false);
                bcur = bnext;
            }
        }

        // ---- bias + swish, C-layout -> A-layout transpose via LDS ----
        #pragma unroll
        for (int t = 0; t < 8; t++) {
            const int col = t * 16 + l16;
            const float b = bm1[col];
            #pragma unroll
            for (int r = 0; r < 8; r++) {
                const int m = r + hi * 8;
                act[wave][m][col]      = (__bf16)swishf(acc0[t][r] + b);
                act[wave][16 + m][col] = (__bf16)swishf(acc1[t][r] + b);
            }
        }
        // act slab is private to this wave; DS ops are in-order per wave.

        // ---- layer 2: [32,128] x [128,128] ----
        v8f acc20[8] = {};
        v8f acc21[8] = {};
        #pragma unroll
        for (int ks = 0; ks < 4; ks++) {
            const int k0 = ks * 32;
            v16bf a0 = load_frag_lds(&act[wave][l16][k0 + hi * 16]);
            v16bf a1 = load_frag_lds(&act[wave][16 + l16][k0 + hi * 16]);

            v16bf bcur = load_frag_lds(&Wm2T[l16][k0 + hi * 16]);
            #pragma unroll
            for (int t = 0; t < 8; t++) {
                v16bf bnext = bcur;
                if (t < 7) bnext = load_frag_lds(&Wm2T[(t + 1) * 16 + l16][k0 + hi * 16]);
                acc20[t] = __builtin_amdgcn_wmma_f32_16x16x32_bf16(
                    false, a0, false, bcur, (short)0, acc20[t], false, false);
                acc21[t] = __builtin_amdgcn_wmma_f32_16x16x32_bf16(
                    false, a1, false, bcur, (short)0, acc21[t], false, false);
                bcur = bnext;
            }
        }

        // ---- bias + atomic scatter-add (segment_sum over receivers) ----
        #pragma unroll
        for (int t = 0; t < 8; t++) {
            const int col = t * 16 + l16;
            const float b = bm2[col];
            #pragma unroll
            for (int r = 0; r < 8; r++) {
                const int m = r + hi * 8;
                const int era = ebase + m;
                const int erb = ebase + 16 + m;
                if (era < E) {
                    atomicAdd(&agg[(size_t)receivers[era] * HDIM + col],
                              acc20[t][r] + b);
                }
                if (erb < E) {
                    atomicAdd(&agg[(size_t)receivers[erb] * HDIM + col],
                              acc21[t][r] + b);
                }
            }
        }
    }
}

// ---------------------------------------------------------------------------
// Node update MLP: out = swish([nodes|agg] @ Wu1 + bu1) @ Wu2 + bu2
// 8 waves/WG, 32 rows per wave.  LDS: 64KB + 32KB + 64KB = 160KB.
// ---------------------------------------------------------------------------
__launch_bounds__(256, 1)
__global__ void node_mlp_kernel(const float* __restrict__ nodes,
                                const float* __restrict__ agg,
                                const float* __restrict__ Wu1,
                                const float* __restrict__ bu1,
                                const float* __restrict__ Wu2,
                                const float* __restrict__ bu2,
                                float*       __restrict__ out,
                                int N, int n_tiles)
{
    __shared__ __align__(32) __bf16 Wu1T[HDIM][K2];       // 65536 B
    __shared__ __align__(32) __bf16 Wu2T[HDIM][HDIM];     // 32768 B
    __shared__ __align__(32) __bf16 act[8][32][HDIM];     // 65536 B

    const int tid = threadIdx.x;
    for (int i = tid; i < K2 * HDIM; i += 256) {
        int k = i >> 7;
        int n = i & 127;
        Wu1T[n][k] = (__bf16)Wu1[i];
    }
    for (int i = tid; i < HDIM * HDIM; i += 256) {
        int k = i >> 7;
        int n = i & 127;
        Wu2T[n][k] = (__bf16)Wu2[i];
    }
    __syncthreads();

    const int lane = tid & 31;
    const int wave = tid >> 5;
    const int hi   = lane >> 4;
    const int l16  = lane & 15;

    for (int tile = blockIdx.x; tile < n_tiles; tile += gridDim.x) {
        const int rbase = tile * 256 + wave * 32;
        const int r0 = rbase + l16;
        const int r1 = rbase + 16 + l16;
        const int rc0 = (r0 < N) ? r0 : (N - 1);
        const int rc1 = (r1 < N) ? r1 : (N - 1);
        const float* p0a = nodes + (size_t)rc0 * FDIM;
        const float* p1a = agg   + (size_t)rc0 * HDIM;
        const float* p0b = nodes + (size_t)rc1 * FDIM;
        const float* p1b = agg   + (size_t)rc1 * HDIM;

        v8f acc0[8] = {};
        v8f acc1[8] = {};

        #pragma unroll
        for (int ks = 0; ks < 8; ks++) {
            const int k0 = ks * 32;
            const int ko = (k0 & 127) + hi * 16;
            const float* ba = (k0 < 128) ? p0a : p1a;
            const float* bb = (k0 < 128) ? p0b : p1b;
            v16bf a0 = load_frag_g(ba, ko);
            v16bf a1 = load_frag_g(bb, ko);

            v16bf bcur = load_frag_lds(&Wu1T[l16][k0 + hi * 16]);
            #pragma unroll
            for (int t = 0; t < 8; t++) {
                v16bf bnext = bcur;
                if (t < 7) bnext = load_frag_lds(&Wu1T[(t + 1) * 16 + l16][k0 + hi * 16]);
                acc0[t] = __builtin_amdgcn_wmma_f32_16x16x32_bf16(
                    false, a0, false, bcur, (short)0, acc0[t], false, false);
                acc1[t] = __builtin_amdgcn_wmma_f32_16x16x32_bf16(
                    false, a1, false, bcur, (short)0, acc1[t], false, false);
                bcur = bnext;
            }
        }

        #pragma unroll
        for (int t = 0; t < 8; t++) {
            const int col = t * 16 + l16;
            const float b = bu1[col];
            #pragma unroll
            for (int r = 0; r < 8; r++) {
                const int m = r + hi * 8;
                act[wave][m][col]      = (__bf16)swishf(acc0[t][r] + b);
                act[wave][16 + m][col] = (__bf16)swishf(acc1[t][r] + b);
            }
        }

        v8f acc20[8] = {};
        v8f acc21[8] = {};
        #pragma unroll
        for (int ks = 0; ks < 4; ks++) {
            const int k0 = ks * 32;
            v16bf a0 = load_frag_lds(&act[wave][l16][k0 + hi * 16]);
            v16bf a1 = load_frag_lds(&act[wave][16 + l16][k0 + hi * 16]);

            v16bf bcur = load_frag_lds(&Wu2T[l16][k0 + hi * 16]);
            #pragma unroll
            for (int t = 0; t < 8; t++) {
                v16bf bnext = bcur;
                if (t < 7) bnext = load_frag_lds(&Wu2T[(t + 1) * 16 + l16][k0 + hi * 16]);
                acc20[t] = __builtin_amdgcn_wmma_f32_16x16x32_bf16(
                    false, a0, false, bcur, (short)0, acc20[t], false, false);
                acc21[t] = __builtin_amdgcn_wmma_f32_16x16x32_bf16(
                    false, a1, false, bcur, (short)0, acc21[t], false, false);
                bcur = bnext;
            }
        }

        #pragma unroll
        for (int t = 0; t < 8; t++) {
            const int col = t * 16 + l16;
            const float b = bu2[col];
            #pragma unroll
            for (int r = 0; r < 8; r++) {
                const int m = r + hi * 8;
                const int ra = rbase + m;
                const int rb = rbase + 16 + m;
                if (ra < N) out[(size_t)ra * HDIM + col] = acc20[t][r] + b;
                if (rb < N) out[(size_t)rb * HDIM + col] = acc21[t][r] + b;
            }
        }
    }
}

extern "C" void kernel_launch(void* const* d_in, const int* in_sizes, int n_in,
                              void* d_out, int out_size, void* d_ws, size_t ws_size,
                              hipStream_t stream) {
    const float* nodes     = (const float*)d_in[0];
    const float* edges     = (const float*)d_in[1];
    const int*   senders   = (const int*)  d_in[2];
    const int*   receivers = (const int*)  d_in[3];
    const float* Wm1       = (const float*)d_in[4];
    const float* bm1       = (const float*)d_in[5];
    const float* Wm2       = (const float*)d_in[6];
    const float* bm2       = (const float*)d_in[7];
    const float* Wu1       = (const float*)d_in[8];
    const float* bu1       = (const float*)d_in[9];
    const float* Wu2       = (const float*)d_in[10];
    const float* bu2       = (const float*)d_in[11];

    const int N = in_sizes[0] / FDIM;   // 50000
    const int E = in_sizes[2];          // 800000
    float* agg = (float*)d_ws;          // [N, H] segment-sum accumulator
    float* out = (float*)d_out;

    hipMemsetAsync(agg, 0, (size_t)N * HDIM * sizeof(float), stream);

    const int tilesE = (E + 255) / 256;
    int gridE = tilesE < 640 ? tilesE : 640;   // persistent WGs
    edge_mlp_kernel<<<gridE, 256, 0, stream>>>(nodes, edges, senders, receivers,
                                               Wm1, bm1, Wm2, bm2, agg, E, tilesE);

    const int tilesN = (N + 255) / 256;
    int gridN = tilesN < 512 ? tilesN : 512;
    node_mlp_kernel<<<gridN, 256, 0, stream>>>(nodes, agg, Wu1, bu1, Wu2, bu2,
                                               out, N, tilesN);
}
